// Att_cnn2d_model_63247688401328
// MI455X (gfx1250) — matrix operand order repaired and checked by reading, not compiled, not run on hardware
//
#include <hip/hip_runtime.h>
#include <hip/hip_bf16.h>
#include <cstdint>

typedef __attribute__((ext_vector_type(16))) __bf16 v16bf;
typedef __attribute__((ext_vector_type(8)))  __bf16 v8bf;
typedef __attribute__((ext_vector_type(8)))  float  v8f;

#define B_SZ    1024
#define T_SZ    32
#define FDIM    64
#define HATT    64
#define NFILT   64
#define OH      31
#define OW      127
#define SPAT    (OH*OW)            // 3937
#define FLATD   (NFILT*SPAT)       // 251968
#define LIN1_IN (FLATD + 24)       // 251992
#define K_PAD   252416             // = 16 * 493 * 32
#define KB_TOT  (K_PAD/32)         // 7888
#define KSPLIT  16
#define KSTEPS  (KB_TOT/KSPLIT)    // 493 (odd)
#define N1      256

// ---------------- attention: inp[b,t,0:64]=X, inp[b,t,64:128]=c ----------------
__global__ __launch_bounds__(256) void attention_kernel(
    const float* __restrict__ X, const float* __restrict__ W,
    const float* __restrict__ U, const float* __restrict__ V,
    float* __restrict__ inp) {
  __shared__ float Xs[T_SZ][FDIM];
  __shared__ float Xw[T_SZ][HATT];
  __shared__ float Xu[T_SZ][HATT];
  __shared__ float Es[T_SZ][T_SZ];
  __shared__ float Vs[HATT];
  const int b = blockIdx.x;
  const int t = threadIdx.x;
  const float* Xb = X + (size_t)b * (T_SZ * FDIM);
  for (int m = 0; m < 8; ++m) ((float*)Xs)[t + 256*m] = Xb[t + 256*m];
  if (t < HATT) Vs[t] = V[t];
  __syncthreads();
  // Xw = X@W, Xu = X@U  (32x64 each)
  for (int m = 0; m < 16; ++m) {
    int idx = (t + 256*m) & 2047;
    int i = idx >> 6, h = idx & 63;
    const float* Wm = (m < 8) ? W : U;
    float s = 0.f;
#pragma unroll 8
    for (int f = 0; f < FDIM; ++f) s += Xs[i][f] * Wm[f*HATT + h];
    if (m < 8) Xw[i][h] = s; else Xu[i][h] = s;
  }
  __syncthreads();
  // e[i][j] = sum_h tanh(Xw[i]+Xu[j]) * V[h]
  for (int m = 0; m < 4; ++m) {
    int p = t + 256*m;
    int i = p >> 5, j = p & 31;
    float s = 0.f;
    for (int h = 0; h < HATT; ++h) {
      float z = Xw[i][h] + Xu[j][h];
      float th = 1.f - 2.f / (__expf(2.f * z) + 1.f);   // tanh
      s += th * Vs[h];
    }
    Es[i][j] = s;
  }
  __syncthreads();
  if (t < T_SZ) {                                        // softmax over j
    float mx = -1e30f;
    for (int j = 0; j < T_SZ; ++j) mx = fmaxf(mx, Es[t][j]);
    float sum = 0.f;
    for (int j = 0; j < T_SZ; ++j) { float ev = __expf(Es[t][j] - mx); Es[t][j] = ev; sum += ev; }
    float inv = 1.f / sum;
    for (int j = 0; j < T_SZ; ++j) Es[t][j] *= inv;
  }
  __syncthreads();
  float* ob = inp + (size_t)b * (T_SZ * 2 * FDIM);
  for (int m = 0; m < 8; ++m) {
    int p = t + 256*m;
    int i = p >> 6, f = p & 63;
    float s = 0.f;
    for (int j = 0; j < T_SZ; ++j) s += Es[i][j] * Xs[j][f];
    s -= Es[i][i] * Xs[i][f];
    ob[i*128 + f]      = Xs[i][f];
    ob[i*128 + 64 + f] = s;
  }
}

// ------- conv(2x2,relu) + flatten + cate embeddings -> x row (bf16, K_PAD) -------
__global__ __launch_bounds__(256) void conv_flat_kernel(
    const float* __restrict__ inp, const float* __restrict__ cw,
    const float* __restrict__ cb, const int* __restrict__ Xc,
    const float* __restrict__ e0, const float* __restrict__ e1,
    const float* __restrict__ e2, __bf16* __restrict__ xq) {
  __shared__ float ins[T_SZ][128];
  __shared__ float w00[NFILT], w01[NFILT], w10[NFILT], w11[NFILT], bb[NFILT];
  const int b = blockIdx.x;
  const int t = threadIdx.x;
  for (int m = 0; m < 16; ++m)
    ((float*)ins)[t + 256*m] = inp[(size_t)b*4096 + t + 256*m];
  if (t < NFILT) {
    w00[t] = cw[t*4+0]; w01[t] = cw[t*4+1];
    w10[t] = cw[t*4+2]; w11[t] = cw[t*4+3];
    bb[t]  = cb[t];
  }
  __syncthreads();
  __bf16* xr = xq + (size_t)b * K_PAD;
  for (int p = t; p < SPAT; p += 256) {
    int i = p / OW, j = p - i*OW;
    float i00 = ins[i][j],   i01 = ins[i][j+1];
    float i10 = ins[i+1][j], i11 = ins[i+1][j+1];
#pragma unroll 8
    for (int f = 0; f < NFILT; ++f) {
      float v = bb[f] + w00[f]*i00 + w01[f]*i01 + w10[f]*i10 + w11[f]*i11;
      xr[f*SPAT + p] = (__bf16)fmaxf(v, 0.f);
    }
  }
  if (t < 24) {
    int c = t >> 3, d = t & 7;
    int id = Xc[b*3 + c];
    const float* tab = (c == 0) ? e0 : (c == 1) ? e1 : e2;
    xr[FLATD + t] = (__bf16)tab[(size_t)id*8 + d];
  }
  for (int k = LIN1_IN + t; k < K_PAD; k += 256) xr[k] = (__bf16)0.0f;
}

// --- w1 (256 x LIN1_IN fp32) -> bf16, pre-swizzled into 32x16 B-fragment layout ---
// wB[ ((kb*256 + n)*2 + h)*16 + j ] = w1[n, kb*32 + (j&8)*2 + h*8 + (j&7)]   (zero-padded)
__global__ __launch_bounds__(256) void convert_w1_kernel(
    const float* __restrict__ w1, __bf16* __restrict__ wB) {
  const size_t total  = (size_t)K_PAD * 256;
  const size_t stride = (size_t)gridDim.x * 256;
  for (size_t idx = (size_t)blockIdx.x*256 + threadIdx.x; idx < total; idx += stride) {
    int    j  = (int)(idx & 15);
    int    h  = (int)((idx >> 4) & 1);
    int    n  = (int)((idx >> 5) & 255);
    size_t kb = idx >> 13;
    int    k  = (int)(kb * 32) + (j & 8)*2 + h*8 + (j & 7);
    float  v  = (k < LIN1_IN) ? w1[(size_t)n*LIN1_IN + k] : 0.0f;
    wB[idx] = (__bf16)v;
  }
}

__device__ __forceinline__ v16bf load_a_frag(const __bf16* p) {
  v8bf lo = *(const v8bf*)p;
  v8bf hi = *(const v8bf*)(p + 16);
  return __builtin_shufflevector(lo, hi, 0,1,2,3,4,5,6,7,8,9,10,11,12,13,14,15);
}
__device__ __forceinline__ v8f wmma_bf16(v16bf a, v16bf b, v8f c) {
  return __builtin_amdgcn_wmma_f32_16x16x32_bf16(false, a, false, b, (short)0, c, false, false);
}

// ------------- bf16 WMMA GEMM: out1p[z] = partial x(1024xK) * w1^T(Kx256) -------------
// grid (8 M, 2 N, 16 Ksplit); block 256 = 8 waves (4 waveM x 2 waveN);
// wave tile 32x64 = 2 A-frags x 4 B-frags = 8 WMMAs per 32-K step.
// Double-buffered with two DISJOINT buffer sets (manual 2x unroll): no register
// copies between buffers -> no v_mov/v_nop hazard filler in the steady state.
__global__ __launch_bounds__(256) void gemm1_kernel(
    const __bf16* __restrict__ x, const __bf16* __restrict__ wB,
    float* __restrict__ out1p) {
  const int lane  = threadIdx.x & 31;
  const int wave  = threadIdx.x >> 5;
  const int waveM = wave >> 1;          // 0..3
  const int waveN = wave & 1;           // 0..1
  const int llo = lane & 15, lhi = lane >> 4;
  const int m_base = blockIdx.x*128 + waveM*32;
  const int n_base = blockIdx.y*128 + waveN*64;
  const int kb0    = blockIdx.z * KSTEPS;

  v8f acc[8];
#pragma unroll
  for (int i = 0; i < 8; ++i) acc[i] = (v8f){};

  // A: row-major x; lane holds row m_base(+16)+llo, K chunks {8*lhi..+7},{16+8*lhi..+7}
  const __bf16* a0p = x + (size_t)(m_base + llo) * K_PAD + (size_t)kb0*32 + 8*lhi;
  const __bf16* a1p = a0p + (size_t)16 * K_PAD;
  // B: pre-swizzled; 16 contiguous bf16 per (kb, col, half); col+16 -> +512 bf16
  const __bf16* bp  = wB + (((size_t)kb0*256 + n_base + llo)*2 + lhi)*16;

  v16bf A0[2], B0[4], A1[2], B1[4];
  // prologue: step 0 -> buffer set 0
  A0[0] = load_a_frag(a0p);
  A0[1] = load_a_frag(a1p);
#pragma unroll
  for (int c = 0; c < 4; ++c) B0[c] = *(const v16bf*)(bp + c*512);

  int s = 0;
  for (; s + 2 < KSTEPS; s += 2) {
    // load step s+1 -> set 1
    A1[0] = load_a_frag(a0p + 32);
    A1[1] = load_a_frag(a1p + 32);
#pragma unroll
    for (int c = 0; c < 4; ++c) B1[c] = *(const v16bf*)(bp + 8192 + c*512);
    // compute step s with set 0
#pragma unroll
    for (int mi = 0; mi < 2; ++mi)
#pragma unroll
      for (int ni = 0; ni < 4; ++ni)
        acc[mi*4+ni] = wmma_bf16(A0[mi], B0[ni], acc[mi*4+ni]);
    // load step s+2 -> set 0
    A0[0] = load_a_frag(a0p + 64);
    A0[1] = load_a_frag(a1p + 64);
#pragma unroll
    for (int c = 0; c < 4; ++c) B0[c] = *(const v16bf*)(bp + 16384 + c*512);
    // compute step s+1 with set 1
#pragma unroll
    for (int mi = 0; mi < 2; ++mi)
#pragma unroll
      for (int ni = 0; ni < 4; ++ni)
        acc[mi*4+ni] = wmma_bf16(A1[mi], B1[ni], acc[mi*4+ni]);
    a0p += 64; a1p += 64; bp += 16384;
  }
  // epilogue: last step (KSTEPS is odd) is in set 0
#pragma unroll
  for (int mi = 0; mi < 2; ++mi)
#pragma unroll
    for (int ni = 0; ni < 4; ++ni)
      acc[mi*4+ni] = wmma_bf16(A0[mi], B0[ni], acc[mi*4+ni]);

  // C/D layout: VGPR r -> row 8*lhi + r, col llo (per 16x16 fragment)
  float* op = out1p + (size_t)blockIdx.z * (B_SZ * N1);
#pragma unroll
  for (int mi = 0; mi < 2; ++mi) {
    const int row0 = m_base + 16*mi + 8*lhi;
#pragma unroll
    for (int ni = 0; ni < 4; ++ni) {
      const int col = n_base + 16*ni + llo;
#pragma unroll
      for (int r = 0; r < 8; ++r)
        op[(size_t)(row0 + r)*N1 + col] = acc[mi*4+ni][r];
    }
  }
}

// ------- deterministic K-split reduction + bias/relu + layer2 + layer3 -------
__global__ __launch_bounds__(256) void tail_kernel(
    const float* __restrict__ out1p, const float* __restrict__ b1,
    const float* __restrict__ w2, const float* __restrict__ b2,
    const float* __restrict__ w3, const float* __restrict__ b3,
    float* __restrict__ out) {
  __shared__ float a1[256];
  __shared__ float h2[32];
  const int b = blockIdx.x;
  const int t = threadIdx.x;
  float s = b1[t];
  for (int z = 0; z < KSPLIT; ++z)                       // fixed order: deterministic
    s += out1p[(size_t)z*(B_SZ*N1) + (size_t)b*N1 + t];
  a1[t] = fmaxf(s, 0.f);
  __syncthreads();
  if (t < 32) {
    float s2 = b2[t];
    for (int o = 0; o < 256; ++o) s2 += a1[o] * w2[t*256 + o];
    h2[t] = fmaxf(s2, 0.f);
  }
  __syncthreads();
  if (t < 2) {
    float s3 = b3[t];
    for (int j = 0; j < 32; ++j) s3 += h2[j] * w3[t*32 + j];
    out[b*2 + t] = s3;
  }
}

extern "C" void kernel_launch(void* const* d_in, const int* in_sizes, int n_in,
                              void* d_out, int out_size, void* d_ws, size_t ws_size,
                              hipStream_t stream) {
  const float* X_nume = (const float*)d_in[0];
  const int*   X_cate = (const int*)  d_in[1];
  const float* W_att  = (const float*)d_in[2];
  const float* U_att  = (const float*)d_in[3];
  const float* V_att  = (const float*)d_in[4];
  const float* conv_w = (const float*)d_in[5];
  const float* conv_b = (const float*)d_in[6];
  const float* emb0   = (const float*)d_in[7];
  const float* emb1   = (const float*)d_in[8];
  const float* emb2   = (const float*)d_in[9];
  const float* w1     = (const float*)d_in[10];
  const float* b1     = (const float*)d_in[11];
  const float* w2     = (const float*)d_in[12];
  const float* b2     = (const float*)d_in[13];
  const float* w3     = (const float*)d_in[14];
  const float* b3     = (const float*)d_in[15];

  // workspace layout (bytes):
  //   [0)            inp      : 1024*32*128*4   = 16,777,216
  //   [16,777,216)   x  bf16  : 1024*K_PAD*2    = 516,947,968
  //   [533,725,184)  wB bf16  : K_PAD*256*2     = 129,236,992
  //   [662,962,176)  out1p    : 16*1024*256*4   = 16,777,216   (total ~680 MB)
  char*   ws    = (char*)d_ws;
  float*  inp   = (float*)ws;
  __bf16* xq    = (__bf16*)(ws + 16777216ull);
  __bf16* wB    = (__bf16*)(ws + 533725184ull);
  float*  out1p = (float*)(ws + 662962176ull);
  float*  out   = (float*)d_out;

  attention_kernel<<<B_SZ, 256, 0, stream>>>(X_nume, W_att, U_att, V_att, inp);
  convert_w1_kernel<<<8192, 256, 0, stream>>>(w1, wB);
  conv_flat_kernel<<<B_SZ, 256, 0, stream>>>(inp, conv_w, conv_b, X_cate,
                                             emb0, emb1, emb2, xq);
  gemm1_kernel<<<dim3(8, 2, 16), 256, 0, stream>>>(xq, wB, out1p);
  tail_kernel<<<B_SZ, 256, 0, stream>>>(out1p, b1, w2, b2, w3, b3, out);
}
